// ourRLIFLayer_60876866453588
// MI455X (gfx1250) — compile-verified, matrix operand order checked by reading, not run on hardware
//
#include <hip/hip_runtime.h>

// ---------------------------------------------------------------------------
// RLIF spiking layer for MI455X (gfx1250, wave32, WMMA, async-to-LDS)
//   B=128, T=256, IN=H=1024
//   Phase 0: one-shot f32->bf16 conversion of x and W (if ws is big enough)
//   Phase 1: Wx = x @ W^T  -> bf16 WMMA, f32 accum, written into d_out,
//            A/B panels staged with GLOBAL_LOAD_ASYNC_TO_LDS, double buffered
//   Phase 2: BN stats + affine fold (scale/shift per column)
//   Phase 3: persistent scan kernel, 256 steps, f16 WMMA for s @ V0,
//            u in registers, async LDS staging, grid-wide atomic barrier
// ---------------------------------------------------------------------------

typedef __attribute__((ext_vector_type(16))) _Float16 v16h;
typedef __attribute__((ext_vector_type(16))) __bf16   v16bf;
typedef __attribute__((ext_vector_type(8)))  float    v8f;

#define BDIM  128
#define TDIM  256
#define KDIM  1024
#define HDIM  1024
#define BT    (BDIM * TDIM)       // 32768 rows of Wx

#define TK    32                  // K-chunk (bf16/f16 WMMA shapes are 16x16x32)
#define LDA   40                  // LDS row stride in 16-bit elems (pad vs 32)

// workspace layout (bytes)
#define OFF_VT     0u                              // f16 V0^T [H][H]      2 MB
#define OFF_SBUF   0x200000u                       // f16 spikes [2][128][1024]
#define OFF_SUMS   (OFF_SBUF + 2u * 128u * 1024u * 2u)
#define OFF_SUMSQ  (OFF_SUMS  + 4096u)
#define OFF_SCALE  (OFF_SUMSQ + 4096u)
#define OFF_SHIFT  (OFF_SCALE + 4096u)
#define OFF_BAR    (OFF_SHIFT + 4096u)
#define OFF_XB     0x300000u                       // bf16 x  [32768][1024] 64 MB
#define OFF_WB     (OFF_XB + 67108864u)            // bf16 W  [1024][1024]   2 MB
#define OFF_END    (OFF_WB + 2097152u)

// ----------------------------- helpers -------------------------------------

__device__ __forceinline__ unsigned short f2bf(float f) {
  // round-to-nearest-even f32 -> bf16 (bit trick)
  unsigned u = __float_as_uint(f);
  u += 0x7FFFu + ((u >> 16) & 1u);
  return (unsigned short)(u >> 16);
}

union FragBF { uint4 q[2]; v16bf v; };
union FragH  { uint4 q[2]; v16h  v; };

// CDNA5 async memory->LDS copies (ASYNCcnt-tracked; no VGPR round trip).
// The LDS operand is produced by ptrtoint of the real shared pointer: the low
// 32 bits of a shared-aperture flat address are the LDS byte offset (ISA
// flat-addressing rules), and the ptrtoint makes the shared array escape so
// the "memory" clobber keeps the subsequent ds_load fragment reads alive.
__device__ __forceinline__ void async_b128(const unsigned short* lds, const void* g) {
  unsigned off = (unsigned)(unsigned long long)lds;
  asm volatile("global_load_async_to_lds_b128 %0, %1, off"
               :: "v"(off), "v"(g) : "memory");
}
__device__ __forceinline__ void async_b32(const unsigned short* lds, const void* g) {
  unsigned off = (unsigned)(unsigned long long)lds;
  asm volatile("global_load_async_to_lds_b32 %0, %1, off"
               :: "v"(off), "v"(g) : "memory");
}
__device__ __forceinline__ void wait_async_3() {
  asm volatile("s_wait_asynccnt 0x3" ::: "memory");
}
__device__ __forceinline__ void wait_async_0() {
  asm volatile("s_wait_asynccnt 0x0" ::: "memory");
}

// ----------------------------- init ----------------------------------------

__global__ __launch_bounds__(256) void init_ws(unsigned* __restrict__ p, int n) {
  int i = blockIdx.x * 256 + threadIdx.x;
  if (i < n) p[i] = 0u;
}

// --------------------- conversions ------------------------------------------

__global__ __launch_bounds__(256) void convert_bf16(const float* __restrict__ s,
                                                    unsigned short* __restrict__ d,
                                                    int n8) {  // n/8 groups
  int i = blockIdx.x * 256 + threadIdx.x;
  if (i < n8) {
    const float4* p = (const float4*)s + (size_t)i * 2;
    float4 f0 = p[0], f1 = p[1];
    uint4 o;
    o.x = f2bf(f0.x) | ((unsigned)f2bf(f0.y) << 16);
    o.y = f2bf(f0.z) | ((unsigned)f2bf(f0.w) << 16);
    o.z = f2bf(f1.x) | ((unsigned)f2bf(f1.y) << 16);
    o.w = f2bf(f1.z) | ((unsigned)f2bf(f1.w) << 16);
    *((uint4*)d + i) = o;
  }
}

__global__ __launch_bounds__(256) void convert_v(const float* __restrict__ V,
                                                 unsigned short* __restrict__ Vt) {
  // Vt[n][k] = f16( k==n ? 0 : V[k][n] );  B-operand wants fixed-n contiguous-k
  int idx = blockIdx.x * 256 + threadIdx.x;       // 1M elements
  int n = idx >> 10, k = idx & 1023;
  float v = (k == n) ? 0.0f : V[(size_t)k * HDIM + n];
  _Float16 h = (_Float16)v;
  Vt[idx] = __builtin_bit_cast(unsigned short, h);
}

// ---------------- Wx GEMM, fast path: bf16 source + async LDS ----------------
// Workgroup tile 128(M) x 64(N), 8 waves in 4x2 grid, wave tile 32x32.
// LDS: double-buffered A(128x32) and B(64x32) bf16 panels.

#define G_AS0 0
#define G_AS1 (128 * LDA)
#define G_BS0 (2 * 128 * LDA)
#define G_BS1 (2 * 128 * LDA + 64 * LDA)
#define G_SMEM (2 * 128 * LDA + 2 * 64 * LDA)      // 15360 halfs = 30720 B

__global__ __launch_bounds__(256) void gemm_wx_bf16(
    const unsigned short* __restrict__ xb,   // bf16 [32768][1024]
    const unsigned short* __restrict__ wb,   // bf16 [1024][1024]
    float* __restrict__ out) {
  __shared__ __align__(16) unsigned short smem[G_SMEM];

  const int tid  = threadIdx.x;
  const int lane = tid & 31;
  const int wave = tid >> 5;
  const int wm   = wave & 3;
  const int wn   = wave >> 2;
  const int m0   = blockIdx.x * 128;
  const int n0   = blockIdx.y * 64;
  const int r    = lane & 15;
  const int hilo = (lane >> 4) * 8;

  // per-thread staging coordinates (3 async instrs per chunk, uniform)
  const int arow = tid >> 1;
  const int acol = (tid & 1) * 16;     // two b128: halfs [acol,+8),[acol+8,+8)
  const int brow = tid >> 2;
  const int bcol = (tid & 3) * 8;      // one b128
  const unsigned short* agp = xb + (size_t)(m0 + arow) * KDIM + acol;
  const unsigned short* bgp = wb + (size_t)(n0 + brow) * KDIM + bcol;

  auto issue = [&](int k0, int buf) {
    unsigned short* ab = &smem[(buf ? G_AS1 : G_AS0) + arow * LDA + acol];
    unsigned short* bb = &smem[(buf ? G_BS1 : G_BS0) + brow * LDA + bcol];
    async_b128(ab,     agp + k0);
    async_b128(ab + 8, agp + k0 + 8);
    async_b128(bb,     bgp + k0);
  };

  v8f acc[2][2];
#pragma unroll
  for (int i = 0; i < 2; ++i)
#pragma unroll
    for (int j = 0; j < 2; ++j) acc[i][j] = (v8f)(0.0f);

  issue(0, 0);
  for (int kb = 0; kb < KDIM / TK; ++kb) {
    const int cur = kb & 1;
    if (kb + 1 < KDIM / TK) {          // prefetch next chunk, then wait cur
      issue((kb + 1) * TK, 1 - cur);
      wait_async_3();
    } else {
      wait_async_0();
    }
    __syncthreads();

    const unsigned as = cur ? G_AS1 : G_AS0;
    const unsigned bs = cur ? G_BS1 : G_BS0;
    FragBF a[2], b[2];
#pragma unroll
    for (int i = 0; i < 2; ++i) {
      const int row = wm * 32 + i * 16 + r;
      a[i].q[0] = *(const uint4*)&smem[as + row * LDA + hilo];
      a[i].q[1] = *(const uint4*)&smem[as + row * LDA + hilo + 16];
    }
#pragma unroll
    for (int j = 0; j < 2; ++j) {
      const int col = wn * 32 + j * 16 + r;
      b[j].q[0] = *(const uint4*)&smem[bs + col * LDA + hilo];
      b[j].q[1] = *(const uint4*)&smem[bs + col * LDA + hilo + 16];
    }
#pragma unroll
    for (int i = 0; i < 2; ++i)
#pragma unroll
      for (int j = 0; j < 2; ++j)
        acc[i][j] = __builtin_amdgcn_wmma_f32_16x16x32_bf16(
            false, a[i].v, false, b[j].v, (short)0, acc[i][j], false, false);
    __syncthreads();
  }

#pragma unroll
  for (int i = 0; i < 2; ++i)
#pragma unroll
    for (int j = 0; j < 2; ++j)
#pragma unroll
      for (int e = 0; e < 8; ++e) {
        const int m = m0 + wm * 32 + i * 16 + e + hilo;
        const int n = n0 + wn * 32 + j * 16 + r;
        out[(size_t)m * HDIM + n] = acc[i][j][e];
      }
}

// ---------------- Wx GEMM, fallback: fused f32->bf16 staging ----------------

__global__ __launch_bounds__(256) void gemm_wx(const float* __restrict__ x,
                                               const float* __restrict__ W,
                                               float* __restrict__ out) {
  __shared__ unsigned short As[128][LDA];
  __shared__ unsigned short Bs[64][LDA];

  const int tid  = threadIdx.x;
  const int lane = tid & 31;
  const int wave = tid >> 5;
  const int wm   = wave & 3;
  const int wn   = wave >> 2;
  const int m0   = blockIdx.x * 128;
  const int n0   = blockIdx.y * 64;
  const int r    = lane & 15;
  const int hilo = (lane >> 4) * 8;

  v8f acc[2][2];
#pragma unroll
  for (int i = 0; i < 2; ++i)
#pragma unroll
    for (int j = 0; j < 2; ++j) acc[i][j] = (v8f)(0.0f);

  for (int k0 = 0; k0 < KDIM; k0 += TK) {
    {
      const int row = tid >> 1;
      const int seg = (tid & 1) * 16;
      const float4* src = (const float4*)(x + (size_t)(m0 + row) * KDIM + k0 + seg);
#pragma unroll
      for (int i = 0; i < 4; ++i) {
        float4 f = src[i];
        uint2 p;
        p.x = f2bf(f.x) | ((unsigned)f2bf(f.y) << 16);
        p.y = f2bf(f.z) | ((unsigned)f2bf(f.w) << 16);
        *(uint2*)&As[row][seg + i * 4] = p;
      }
    }
    {
      const int row = tid >> 2;
      const int seg = (tid & 3) * 8;
      const float4* src = (const float4*)(W + (size_t)(n0 + row) * KDIM + k0 + seg);
#pragma unroll
      for (int i = 0; i < 2; ++i) {
        float4 f = src[i];
        uint2 p;
        p.x = f2bf(f.x) | ((unsigned)f2bf(f.y) << 16);
        p.y = f2bf(f.z) | ((unsigned)f2bf(f.w) << 16);
        *(uint2*)&Bs[row][seg + i * 4] = p;
      }
    }
    __syncthreads();

    FragBF a[2], b[2];
#pragma unroll
    for (int i = 0; i < 2; ++i) {
      const int row = wm * 32 + i * 16 + r;
      a[i].q[0] = *(const uint4*)&As[row][hilo];
      a[i].q[1] = *(const uint4*)&As[row][hilo + 16];
    }
#pragma unroll
    for (int j = 0; j < 2; ++j) {
      const int col = wn * 32 + j * 16 + r;
      b[j].q[0] = *(const uint4*)&Bs[col][hilo];
      b[j].q[1] = *(const uint4*)&Bs[col][hilo + 16];
    }
#pragma unroll
    for (int i = 0; i < 2; ++i)
#pragma unroll
      for (int j = 0; j < 2; ++j)
        acc[i][j] = __builtin_amdgcn_wmma_f32_16x16x32_bf16(
            false, a[i].v, false, b[j].v, (short)0, acc[i][j], false, false);
    __syncthreads();
  }

#pragma unroll
  for (int i = 0; i < 2; ++i)
#pragma unroll
    for (int j = 0; j < 2; ++j)
#pragma unroll
      for (int e = 0; e < 8; ++e) {
        const int m = m0 + wm * 32 + i * 16 + e + hilo;
        const int n = n0 + wn * 32 + j * 16 + r;
        out[(size_t)m * HDIM + n] = acc[i][j][e];
      }
}

// ----------------------------- BN statistics --------------------------------

__global__ __launch_bounds__(256) void bn_stats(const float* __restrict__ wx,
                                                float* __restrict__ sums,
                                                float* __restrict__ sumsq) {
  const int col  = (blockIdx.x & 3) * 256 + threadIdx.x;
  const int row0 = (blockIdx.x >> 2) * 1024;
  float s1 = 0.f, s2 = 0.f;
  for (int rr = 0; rr < 1024; ++rr) {
    float v = wx[(size_t)(row0 + rr) * HDIM + col];
    s1 += v;
    s2 += v * v;
  }
  atomicAdd(&sums[col], s1);
  atomicAdd(&sumsq[col], s2);
}

__global__ __launch_bounds__(256) void bn_finalize(const float* __restrict__ sums,
                                                   const float* __restrict__ sumsq,
                                                   const float* __restrict__ gamma,
                                                   const float* __restrict__ beta,
                                                   float* __restrict__ scale,
                                                   float* __restrict__ shift) {
  const int h = blockIdx.x * 256 + threadIdx.x;
  if (h < HDIM) {
    const float inv  = 1.0f / (float)BT;
    const float mean = sums[h] * inv;
    const float var  = sumsq[h] * inv - mean * mean;
    const float sc   = gamma[h] * rsqrtf(var + 1e-5f);
    scale[h] = sc;
    shift[h] = beta[h] - mean * sc;
  }
}

// ------------------------- persistent recurrent scan ------------------------

__device__ __forceinline__ void grid_barrier(unsigned* bar, unsigned nblocks) {
  __syncthreads();
  if (threadIdx.x == 0) {
    volatile unsigned* vgen = bar + 32;
    const unsigned g = *vgen;
    __threadfence();
    const unsigned arrived = atomicAdd(&bar[0], 1u);
    if (arrived == nblocks - 1u) {
      bar[0] = 0u;
      __threadfence();
      atomicAdd(&bar[32], 1u);
    } else {
      while (*vgen == g) __builtin_amdgcn_s_sleep(1);
    }
    __threadfence();
  }
  __syncthreads();
}

// LDS map for scan: double-buffered spike panel (128x32) and Vt panel (16x32)
#define S_AS0 0
#define S_AS1 (128 * LDA)
#define S_BS0 (2 * 128 * LDA)
#define S_BS1 (2 * 128 * LDA + 16 * LDA)
#define S_SMEM (2 * 128 * LDA + 2 * 16 * LDA)      // 11520 halfs = 23040 B

// 64 WGs x 256 threads; WG owns all 128 batch rows x 16 columns. Wave w
// handles batch rows [16w,16w+16). Membrane u lives in registers in the
// WMMA C layout; Wx (in d_out) is read once per (b,t,h), overwritten by spike.
__global__ __launch_bounds__(256) void scan_kernel(
    float* __restrict__ wx_out,             // [128][256][1024]
    const unsigned short* __restrict__ Vt,  // f16 V0^T [1024][1024]
    unsigned short* __restrict__ sbuf,      // f16 [2][128][1024]
    const float* __restrict__ scale,
    const float* __restrict__ shift,
    unsigned* __restrict__ bar) {
  __shared__ __align__(16) unsigned short smem[S_SMEM];

  const int tid  = threadIdx.x;
  const int lane = tid & 31;
  const int wave = tid >> 5;                // batch rows wave*16
  const int n0   = blockIdx.x * 16;
  const int r    = lane & 15;
  const int hilo = (lane >> 4) * 8;

  const int   h  = n0 + r;
  const float sc = scale[h];
  const float sh = shift[h];

  // staging coords: 3 async instrs per chunk, uniform over all threads
  const int arow = tid >> 1;
  const int acol = (tid & 1) * 16;          // two b128
  const int brow = tid >> 4;                // 16 rows x 16 threads
  const int bcol = (tid & 15) * 2;          // one b32 (2 halfs)
  const unsigned short* bgp = Vt + (size_t)(n0 + brow) * HDIM + bcol;

  float u[8];
#pragma unroll
  for (int e = 0; e < 8; ++e) u[e] = 0.0f;

  for (int t = 0; t < TDIM; ++t) {
    const unsigned short* sprev = sbuf + (size_t)(t & 1) * BDIM * HDIM;
    unsigned short*       snext = sbuf + (size_t)((t + 1) & 1) * BDIM * HDIM;
    const unsigned short* agp   = sprev + (size_t)arow * HDIM + acol;

    auto issue = [&](int k0, int buf) {
      unsigned short* ab = &smem[(buf ? S_AS1 : S_AS0) + arow * LDA + acol];
      unsigned short* bb = &smem[(buf ? S_BS1 : S_BS0) + brow * LDA + bcol];
      async_b128(ab,     agp + k0);
      async_b128(ab + 8, agp + k0 + 8);
      async_b32(bb,      bgp + k0);
    };

    v8f acc = (v8f)(0.0f);
    issue(0, 0);
    for (int kb = 0; kb < HDIM / TK; ++kb) {
      const int cur = kb & 1;
      if (kb + 1 < HDIM / TK) {
        issue((kb + 1) * TK, 1 - cur);
        wait_async_3();
      } else {
        wait_async_0();
      }
      __syncthreads();

      const unsigned as = cur ? S_AS1 : S_AS0;
      const unsigned bs = cur ? S_BS1 : S_BS0;
      FragH a, b;
      const int ar = wave * 16 + r;
      a.q[0] = *(const uint4*)&smem[as + ar * LDA + hilo];
      a.q[1] = *(const uint4*)&smem[as + ar * LDA + hilo + 16];
      b.q[0] = *(const uint4*)&smem[bs + r * LDA + hilo];
      b.q[1] = *(const uint4*)&smem[bs + r * LDA + hilo + 16];

      acc = __builtin_amdgcn_wmma_f32_16x16x32_f16(
          false, a.v, false, b.v, (short)0, acc, false, false);
      __syncthreads();
    }

    // LIF update on the 16x16 accumulator this wave owns
#pragma unroll
    for (int e = 0; e < 8; ++e) {
      const int    bi  = wave * 16 + e + hilo;                 // batch row
      const size_t idx = ((size_t)bi * TDIM + t) * HDIM + h;
      const float  wxn = wx_out[idx] * sc + sh;                // folded BN
      const float  un  = 0.5f * u[e] + wxn + acc[e];           // TAU=0.5
      const bool   sp  = un > 1.0f;                            // THRESHOLD=1
      wx_out[idx] = sp ? 1.0f : 0.0f;                          // spike out
      u[e]        = sp ? 0.0f : un;                            // hard reset
      snext[(size_t)bi * HDIM + h] =
          sp ? (unsigned short)0x3C00 : (unsigned short)0;     // f16 spike
    }

    grid_barrier(bar, gridDim.x);
  }
}

// ------------------------------- launcher -----------------------------------

extern "C" void kernel_launch(void* const* d_in, const int* in_sizes, int n_in,
                              void* d_out, int out_size, void* d_ws, size_t ws_size,
                              hipStream_t stream) {
  (void)in_sizes; (void)n_in; (void)out_size;

  const float* x     = (const float*)d_in[1];   // [128,256,1024]
  const float* W     = (const float*)d_in[2];   // [1024,1024]
  const float* V     = (const float*)d_in[3];   // [1024,1024]
  const float* gamma = (const float*)d_in[4];
  const float* beta  = (const float*)d_in[5];

  char* ws = (char*)d_ws;
  unsigned short* Vt    = (unsigned short*)(ws + OFF_VT);
  unsigned short* sbuf  = (unsigned short*)(ws + OFF_SBUF);
  float*          sums  = (float*)(ws + OFF_SUMS);
  float*          sumsq = (float*)(ws + OFF_SUMSQ);
  float*          scl   = (float*)(ws + OFF_SCALE);
  float*          shf   = (float*)(ws + OFF_SHIFT);
  unsigned*       bar   = (unsigned*)(ws + OFF_BAR);
  unsigned short* xb    = (unsigned short*)(ws + OFF_XB);
  unsigned short* wb    = (unsigned short*)(ws + OFF_WB);

  float* out = (float*)d_out;

  // zero spike buffers, BN accumulators, barrier state (harness poisons ws)
  const int zero_u32 = (int)((OFF_BAR + 256u - OFF_SBUF) / 4u);
  init_ws<<<(zero_u32 + 255) / 256, 256, 0, stream>>>(
      (unsigned*)(ws + OFF_SBUF), zero_u32);

  convert_v<<<(HDIM * HDIM) / 256, 256, 0, stream>>>(V, Vt);

  if (ws_size >= (size_t)OFF_END) {
    // fast path: pre-convert to bf16 once, async-LDS double-buffered GEMM
    convert_bf16<<<(BT * KDIM / 8 + 255) / 256, 256, 0, stream>>>(
        x, xb, BT * KDIM / 8);
    convert_bf16<<<(HDIM * KDIM / 8 + 255) / 256, 256, 0, stream>>>(
        W, wb, HDIM * KDIM / 8);
    gemm_wx_bf16<<<dim3(BT / 128, HDIM / 64), 256, 0, stream>>>(xb, wb, out);
  } else {
    gemm_wx<<<dim3(BT / 128, HDIM / 64), 256, 0, stream>>>(x, W, out);
  }

  bn_stats<<<128, 256, 0, stream>>>(out, sums, sumsq);
  bn_finalize<<<4, 256, 0, stream>>>(sums, sumsq, gamma, beta, scl, shf);

  scan_kernel<<<HDIM / 16, 256, 0, stream>>>(out, Vt, sbuf, scl, shf, bar);
}
